// Pct_11742440588127
// MI455X (gfx1250) — matrix operand order repaired and checked by reading, not compiled, not run on hardware
//
#include <hip/hip_runtime.h>

// PCT encoder for MI455X (gfx1250, wave32). All matmuls via v_wmma_f32_16x16x32_f16
// with f16 LDS staging using the CDNA5 16-bit A(16x32)/B(32x16) lane layouts.
// GEMM uses double-buffered LDS (1 barrier/K-step), b128 global loads + b64 LDS
// stores on the (dominant) fully-in-bounds tile path, global_prefetch for k+2.
// Side kernels: BN stats/apply, FPS, KNN, gathers, softmax, offset-attn norm, max-pools.

typedef _Float16 half_t;
typedef __attribute__((ext_vector_type(4)))  _Float16 v4h;
typedef __attribute__((ext_vector_type(16))) _Float16 v16h;
typedef __attribute__((ext_vector_type(8)))  float    v8f;

#define LDS_STRIDE 40  // halves per LDS tile row (80 B: keeps v4h stores 8B-aligned)

// ---------------------------------------------------------------------------
// WMMA fragment loaders (layouts per CDNA5 ISA §7.12.2)
// A 16x32 f16: lane L -> row M=L%16, half h=L/16; VGPR v holds K pair at
//   k0 = 2*(v&3) + 16*(v>>2) + 8*h
// B 32x16 f16: lane L -> col N=L%16, half h=L/16; VGPR v holds K pair at
//   k0 = 2*(v&3) + 16*(v>>2) + 8*h
// ---------------------------------------------------------------------------
__device__ __forceinline__ v16h frag_a(const half_t* rowBase, int stride, int lane) {
  int r = lane & 15, h = lane >> 4;
  const half_t* p = rowBase + r * stride;
  v16h f;
#pragma unroll
  for (int v = 0; v < 8; ++v) {
    int k0 = ((v & 3) << 1) + ((v >> 2) << 4) + (h << 3);
    f[2 * v]     = p[k0];
    f[2 * v + 1] = p[k0 + 1];
  }
  return f;
}

__device__ __forceinline__ v16h frag_b(const half_t* rowBase, int stride, int lane) {
  int r = lane & 15, h = lane >> 4;
  const half_t* p = rowBase + r * stride;
  v16h f;
#pragma unroll
  for (int v = 0; v < 8; ++v) {
    int k0 = ((v & 3) << 1) + ((v >> 2) << 4) + (h << 3);
    f[2 * v]     = p[k0];
    f[2 * v + 1] = p[k0 + 1];
  }
  return f;
}

// ---------------------------------------------------------------------------
// Generic batched GEMM:  C[b] (MxN) = A[b] (MxK) * B[b] (KxN)  (+bias[m])
//   A: aT==0 -> A[m*lda + k];  aT==1 -> A[k*lda + m].  aBS==0 -> shared weight.
//   B: row-major B[k*ldb + n], batch stride bBS.
// 4 waves / 128 threads; each wave computes a 32x32 sub-tile (2x2 WMMA frags)
// of a 64x64 block tile. K loop in 32-wide slabs, double-buffered in LDS.
// ---------------------------------------------------------------------------
__global__ __launch_bounds__(128) void gemm_wmma(
    const float* __restrict__ A, long aBS, int lda, int aT,
    const float* __restrict__ Bm, long bBS, int ldb,
    float* __restrict__ C, long cBS, int ldc,
    int M, int N, int Kd, const float* __restrict__ bias)
{
  __shared__ __align__(16) half_t As[2][64][LDS_STRIDE];
  __shared__ __align__(16) half_t Bs[2][64][LDS_STRIDE];

  const int b     = blockIdx.z;
  const int nTile = blockIdx.x << 6;
  const int mTile = blockIdx.y << 6;
  const float* Ab = A  + (long)b * aBS;
  const float* Bb = Bm + (long)b * bBS;
  float*       Cb = C  + (long)b * cBS;

  const int tid  = threadIdx.x;
  const int wave = tid >> 5;
  const int lane = tid & 31;
  const int wm   = (wave >> 1) << 5;   // wave sub-tile row origin (0 or 32)
  const int wn   = (wave & 1) << 5;    // wave sub-tile col origin (0 or 32)

  // Stage one 64x32 A slab and one 32x64 B slab (transposed) into LDS buffer `buf`.
  auto stage = [&](int buf, int kk) {
    const bool fullK = (kk + 32 <= Kd);
    // ---- A tile ----
    if (!aT) {
      if (fullK && (mTile + 64 <= M)) {            // fast path: b128 loads, b64 stores
#pragma unroll
        for (int v = 0; v < 4; ++v) {
          int idx = tid + (v << 7);                // 0..511
          int m = idx >> 3, k4 = (idx & 7) << 2;   // 8 float4 per row
          float4 f = *(const float4*)&Ab[(long)(mTile + m) * lda + kk + k4];
          v4h h; h[0] = (half_t)f.x; h[1] = (half_t)f.y;
                 h[2] = (half_t)f.z; h[3] = (half_t)f.w;
          *(v4h*)&As[buf][m][k4] = h;
        }
      } else {                                     // guarded path (conv1 K=3 only)
        for (int i = tid; i < 2048; i += 128) {
          int m = i >> 5, k = i & 31;
          int gm = mTile + m, gk = kk + k;
          float v = (gm < M && gk < Kd) ? Ab[(long)gm * lda + gk] : 0.0f;
          As[buf][m][k] = (half_t)v;
        }
      }
    } else {
      if (fullK && (mTile + 64 <= M)) {            // fast path: b128 loads along m
#pragma unroll
        for (int v = 0; v < 4; ++v) {
          int idx = tid + (v << 7);
          int k = idx >> 4, m4 = (idx & 15) << 2;  // 16 float4 per k-row
          float4 f = *(const float4*)&Ab[(long)(kk + k) * lda + mTile + m4];
          As[buf][m4 + 0][k] = (half_t)f.x;
          As[buf][m4 + 1][k] = (half_t)f.y;
          As[buf][m4 + 2][k] = (half_t)f.z;
          As[buf][m4 + 3][k] = (half_t)f.w;
        }
      } else {
        for (int i = tid; i < 2048; i += 128) {
          int m = i & 63, k = i >> 6;
          int gm = mTile + m, gk = kk + k;
          float v = (gm < M && gk < Kd) ? Ab[(long)gk * lda + gm] : 0.0f;
          As[buf][m][k] = (half_t)v;
        }
      }
    }
    // ---- B tile (stored transposed: Bs[n][k]) ----
    if (fullK && (nTile + 64 <= N)) {
#pragma unroll
      for (int v = 0; v < 4; ++v) {
        int idx = tid + (v << 7);
        int k = idx >> 4, n4 = (idx & 15) << 2;
        float4 f = *(const float4*)&Bb[(long)(kk + k) * ldb + nTile + n4];
        Bs[buf][n4 + 0][k] = (half_t)f.x;
        Bs[buf][n4 + 1][k] = (half_t)f.y;
        Bs[buf][n4 + 2][k] = (half_t)f.z;
        Bs[buf][n4 + 3][k] = (half_t)f.w;
      }
    } else {
      for (int i = tid; i < 2048; i += 128) {
        int n = i & 63, k = i >> 6;
        int gn = nTile + n, gk = kk + k;
        float v = (gn < N && gk < Kd) ? Bb[(long)gk * ldb + gn] : 0.0f;
        Bs[buf][n][k] = (half_t)v;
      }
    }
  };

  v8f acc0 = {}; v8f acc1 = {}; v8f acc2 = {}; v8f acc3 = {};
  const int ksteps = (Kd + 31) >> 5;

  stage(0, 0);
  __syncthreads();

  for (int ks = 0; ks < ksteps; ++ks) {
    const int cur = ks & 1;
    if (ks + 1 < ksteps) {
      stage(cur ^ 1, (ks + 1) << 5);               // overlap next slab with WMMAs
      int kk2 = (ks + 2) << 5;                     // prefetch slab after next
      if (kk2 < Kd && nTile + 64 <= N)
        __builtin_prefetch(&Bb[(long)(kk2 + (tid >> 6)) * ldb + nTile + (tid & 63)], 0, 1);
    }

    v16h a0 = frag_a(&As[cur][wm][0],      LDS_STRIDE, lane);
    v16h a1 = frag_a(&As[cur][wm + 16][0], LDS_STRIDE, lane);
    v16h b0 = frag_b(&Bs[cur][wn][0],      LDS_STRIDE, lane);
    v16h b1 = frag_b(&Bs[cur][wn + 16][0], LDS_STRIDE, lane);

    acc0 = __builtin_amdgcn_wmma_f32_16x16x32_f16(false, a0, false, b0, (short)0, acc0, false, false);
    acc1 = __builtin_amdgcn_wmma_f32_16x16x32_f16(false, a0, false, b1, (short)0, acc1, false, false);
    acc2 = __builtin_amdgcn_wmma_f32_16x16x32_f16(false, a1, false, b0, (short)0, acc2, false, false);
    acc3 = __builtin_amdgcn_wmma_f32_16x16x32_f16(false, a1, false, b1, (short)0, acc3, false, false);
    __syncthreads();
  }

  // ---- epilogue: C layout lane L -> n = L%16; VGPR j -> m = j + 8*(L/16) ----
  const int r = lane & 15, h = lane >> 4;
  v8f accs[4] = {acc0, acc1, acc2, acc3};
#pragma unroll
  for (int t = 0; t < 4; ++t) {
    int mBase = mTile + wm + ((t >> 1) << 4);
    int nBase = nTile + wn + ((t & 1) << 4);
    int n = nBase + r;
#pragma unroll
    for (int j = 0; j < 8; ++j) {
      int m = mBase + j + (h << 3);
      if (m < M && n < N) {
        float v = accs[t][j];
        if (bias) v += bias[m];
        Cb[(long)m * ldc + n] = v;
      }
    }
  }
}

// ---------------------------------------------------------------------------
// BatchNorm (training mode): stats over (batch, length) per channel.
// ---------------------------------------------------------------------------
__global__ void zero_f32(float* __restrict__ p, long n) {
  long i = (long)blockIdx.x * 256 + threadIdx.x;
  if (i < n) p[i] = 0.0f;
}

__global__ __launch_bounds__(256) void bn_partial(
    const float* __restrict__ Y, int Cc, long Nn, long total, int slabs,
    float* __restrict__ sums)
{
  __shared__ float ss[256], sq[256];
  const int c = blockIdx.x;
  const int slab = blockIdx.y;
  long per = (total + slabs - 1) / slabs;
  long lo = (long)slab * per;
  long hi = lo + per; if (hi > total) hi = total;
  float s = 0.0f, q = 0.0f;
  for (long i = lo + threadIdx.x; i < hi; i += 256) {
    long b = i / Nn, n = i - b * Nn;
    float v = Y[(b * Cc + c) * Nn + n];
    s += v; q += v * v;
  }
  ss[threadIdx.x] = s; sq[threadIdx.x] = q;
  __syncthreads();
  for (int st = 128; st > 0; st >>= 1) {
    if (threadIdx.x < st) {
      ss[threadIdx.x] += ss[threadIdx.x + st];
      sq[threadIdx.x] += sq[threadIdx.x + st];
    }
    __syncthreads();
  }
  if (threadIdx.x == 0) {
    atomicAdd(&sums[c], ss[0]);
    atomicAdd(&sums[Cc + c], sq[0]);
  }
}

__global__ void bn_finalize(const float* __restrict__ sums, int Cc, float denom,
                            float eps, float* __restrict__ mean, float* __restrict__ rstd)
{
  int c = blockIdx.x * 256 + threadIdx.x;
  if (c >= Cc) return;
  float m = sums[c] / denom;
  float var = sums[Cc + c] / denom - m * m;
  mean[c] = m;
  rstd[c] = rsqrtf(var + eps);
}

// act: 0=none, 1=ReLU, 2=LeakyReLU(0.2). Out = act(bn(Y)) [+ res].
__global__ void bn_apply(const float* __restrict__ Y, const float* __restrict__ mean,
                         const float* __restrict__ rstd, const float* __restrict__ g,
                         const float* __restrict__ bt, const float* __restrict__ res,
                         long resBS, float* __restrict__ Out, long outBS,
                         int Cc, long Nn, long total, int act)
{
  long i = (long)blockIdx.x * 256 + threadIdx.x;
  if (i >= total) return;
  long perB = (long)Cc * Nn;
  long b = i / perB;
  long rem = i - b * perB;
  int c = (int)(rem / Nn);
  float v = g[c] * (Y[i] - mean[c]) * rstd[c] + bt[c];
  if (act == 1) v = v > 0.0f ? v : 0.0f;
  else if (act == 2) v = v > 0.0f ? v : 0.2f * v;
  if (res) v += res[b * resBS + rem];
  Out[b * outBS + rem] = v;
}

// ---------------------------------------------------------------------------
// Farthest point sampling: one workgroup per batch, xyz channel-major [B,3,Nn].
// ---------------------------------------------------------------------------
template <int P>
__global__ __launch_bounds__(256) void fps_kernel(
    const float* __restrict__ xyz, int Nn, int npoint, int* __restrict__ out)
{
  __shared__ float sd[256];
  __shared__ int   si[256];
  const int b = blockIdx.x, tid = threadIdx.x;
  const float* X = xyz + (long)b * 3 * Nn;
  float px[P], py[P], pz[P], dmin[P];
#pragma unroll
  for (int j = 0; j < P; ++j) {
    int n = tid + (j << 8);
    px[j] = X[n]; py[j] = X[Nn + n]; pz[j] = X[2 * Nn + n];
    dmin[j] = 1e10f;
  }
  int far = 0;
  for (int it = 0; it < npoint; ++it) {
    if (tid == 0) out[(long)b * npoint + it] = far;
    float cx = X[far], cy = X[Nn + far], cz = X[2 * Nn + far];
    float best = -1.0f; int bi = 0x7fffffff;
#pragma unroll
    for (int j = 0; j < P; ++j) {
      int n = tid + (j << 8);
      float dx = px[j] - cx, dy = py[j] - cy, dz = pz[j] - cz;
      float d = dx * dx + dy * dy + dz * dz;
      if (d < dmin[j]) dmin[j] = d;
      if (dmin[j] > best || (dmin[j] == best && n < bi)) { best = dmin[j]; bi = n; }
    }
    sd[tid] = best; si[tid] = bi;
    __syncthreads();
    for (int st = 128; st > 0; st >>= 1) {
      if (tid < st) {
        float od = sd[tid + st]; int oi = si[tid + st];
        if (od > sd[tid] || (od == sd[tid] && oi < si[tid])) { sd[tid] = od; si[tid] = oi; }
      }
      __syncthreads();
    }
    far = si[0];
    __syncthreads();
  }
}

// ---------------------------------------------------------------------------
// KNN (K=32 nearest) via iterative argmin in LDS. Grid (S, B).
// ---------------------------------------------------------------------------
template <int NMAX>
__global__ __launch_bounds__(256) void knn_kernel(
    const float* __restrict__ xyz, const int* __restrict__ fpsIdx,
    int Nn, int S, int* __restrict__ knnIdx)
{
  __shared__ float d[NMAX];
  __shared__ float sd[256];
  __shared__ int   si[256];
  const int b = blockIdx.y, s = blockIdx.x, tid = threadIdx.x;
  const float* X = xyz + (long)b * 3 * Nn;
  const int ctr = fpsIdx[(long)b * S + s];
  const float cx = X[ctr], cy = X[Nn + ctr], cz = X[2 * Nn + ctr];
  for (int n = tid; n < Nn; n += 256) {
    float dx = X[n] - cx, dy = X[Nn + n] - cy, dz = X[2 * Nn + n] - cz;
    d[n] = dx * dx + dy * dy + dz * dz;
  }
  __syncthreads();
  for (int kk = 0; kk < 32; ++kk) {
    float best = 1e30f; int bi = 0x7fffffff;
    for (int n = tid; n < Nn; n += 256)
      if (d[n] < best || (d[n] == best && n < bi)) { best = d[n]; bi = n; }
    sd[tid] = best; si[tid] = bi;
    __syncthreads();
    for (int st = 128; st > 0; st >>= 1) {
      if (tid < st) {
        float od = sd[tid + st]; int oi = si[tid + st];
        if (od < sd[tid] || (od == sd[tid] && oi < si[tid])) { sd[tid] = od; si[tid] = oi; }
      }
      __syncthreads();
    }
    if (tid == 0) {
      knnIdx[((long)b * S + s) * 32 + kk] = si[0];
      d[si[0]] = 1e30f;
    }
    __syncthreads();
  }
}

// ---------------------------------------------------------------------------
// Gathers / pooling / elementwise
// ---------------------------------------------------------------------------
__global__ void gather_xyz(const float* __restrict__ xyz, const int* __restrict__ fpsIdx,
                           int Nn, int S, long total, float* __restrict__ out)
{
  long i = (long)blockIdx.x * 256 + threadIdx.x;
  if (i >= total) return;
  int s = (int)(i % S); long r = i / S;
  int c = (int)(r % 3); int b = (int)(r / 3);
  out[i] = xyz[((long)b * 3 + c) * Nn + fpsIdx[(long)b * S + s]];
}

// Xg[b][c][col]     = pts[b][c][knn] - pts[b][c][center]
// Xg[b][D+c][col]   = pts[b][c][center],  col = s*32+k
__global__ void gather_group(const float* __restrict__ pts, const int* __restrict__ fpsIdx,
                             const int* __restrict__ knnIdx, int Nn, int S, int D,
                             long total, float* __restrict__ Xg)
{
  long i = (long)blockIdx.x * 256 + threadIdx.x;
  if (i >= total) return;
  long cols = (long)S * 32;
  int col = (int)(i % cols); long r = i / cols;
  int c = (int)(r % D); int b = (int)(r / D);
  int s = col >> 5, k = col & 31;
  int pi = fpsIdx[(long)b * S + s];
  int ci = knnIdx[((long)b * S + s) * 32 + k];
  float ctr = pts[((long)b * D + c) * Nn + pi];
  float gv  = pts[((long)b * D + c) * Nn + ci];
  long base = (long)b * (2 * D) * cols;
  Xg[base + (long)c * cols + col]       = gv - ctr;
  Xg[base + (long)(D + c) * cols + col] = ctr;
}

__global__ void max_over_k(const float* __restrict__ Y, int S, int Cc,
                           long outBS, long total, float* __restrict__ out)
{
  long i = (long)blockIdx.x * 256 + threadIdx.x;
  if (i >= total) return;
  int s = (int)(i % S); long r = i / S;
  int c = (int)(r % Cc); int b = (int)(r / Cc);
  const float* p = Y + ((long)b * Cc + c) * ((long)S * 32) + (long)s * 32;
  float m = p[0];
#pragma unroll
  for (int k = 1; k < 32; ++k) m = fmaxf(m, p[k]);
  out[(long)b * outBS + (long)c * S + s] = m;
}

__global__ __launch_bounds__(256) void softmax_rows(float* __restrict__ E, int Mcols)
{
  __shared__ float red[256];
  const int row = blockIdx.x, tid = threadIdx.x;
  float* p = E + (long)row * Mcols;
  float v = p[tid];
  red[tid] = v; __syncthreads();
  for (int st = 128; st > 0; st >>= 1) {
    if (tid < st) red[tid] = fmaxf(red[tid], red[tid + st]);
    __syncthreads();
  }
  float mx = red[0]; __syncthreads();
  float e = expf(v - mx);
  red[tid] = e; __syncthreads();
  for (int st = 128; st > 0; st >>= 1) {
    if (tid < st) red[tid] += red[tid + st];
    __syncthreads();
  }
  p[tid] = e / red[0];
}

// offset-attention double norm: att /= (1e-9 + column-sum over n)
__global__ __launch_bounds__(256) void att_colnorm(float* __restrict__ Att, int Mm)
{
  const int b = blockIdx.x, m = threadIdx.x;
  float* base = Att + (long)b * Mm * Mm;
  float s = 0.0f;
  for (int n = 0; n < Mm; ++n) s += base[(long)n * Mm + m];
  float inv = 1.0f / (1e-9f + s);
  for (int n = 0; n < Mm; ++n) base[(long)n * Mm + m] *= inv;
}

// o[i] = a[b*aBS + r] - b2[i]  (z = x - x_r; x may be a strided cat slice)
__global__ void ew_sub(const float* __restrict__ a, long aBS,
                       const float* __restrict__ b2, long perB, int Bn,
                       float* __restrict__ o)
{
  long i = (long)blockIdx.x * 256 + threadIdx.x;
  if (i >= (long)Bn * perB) return;
  long b = i / perB, r = i - b * perB;
  o[i] = a[b * aBS + r] - b2[i];
}

__global__ void max_over_n(const float* __restrict__ Y, int Nn, long total,
                           float* __restrict__ out)
{
  long i = (long)blockIdx.x * 256 + threadIdx.x;
  if (i >= total) return;
  const float* p = Y + i * Nn;
  float m = p[0];
  for (int n = 1; n < Nn; ++n) m = fmaxf(m, p[n]);
  out[i] = m;
}

// ---------------------------------------------------------------------------
// Host orchestration
// ---------------------------------------------------------------------------
extern "C" void kernel_launch(void* const* d_in, const int* in_sizes, int n_in,
                              void* d_out, int out_size, void* d_ws, size_t ws_size,
                              hipStream_t stream)
{
  (void)in_sizes; (void)n_in; (void)out_size; (void)ws_size;

  constexpr int Bn = 64, Np = 1024, S0 = 512, S1 = 256;

  const float* x       = (const float*)d_in[0];
  const float* conv1_w = (const float*)d_in[1];
  const float* bn1_g   = (const float*)d_in[2];
  const float* bn1_b   = (const float*)d_in[3];
  const float* conv2_w = (const float*)d_in[4];
  const float* bn2_g   = (const float*)d_in[5];
  const float* bn2_b   = (const float*)d_in[6];
  // local0: 7..12  local1: 13..18  (w1, bn1_g, bn1_b, w2, bn2_g, bn2_b)
  const float* l0w1 = (const float*)d_in[7],  *l0g1 = (const float*)d_in[8],  *l0b1 = (const float*)d_in[9];
  const float* l0w2 = (const float*)d_in[10], *l0g2 = (const float*)d_in[11], *l0b2 = (const float*)d_in[12];
  const float* l1w1 = (const float*)d_in[13], *l1g1 = (const float*)d_in[14], *l1b1 = (const float*)d_in[15];
  const float* l1w2 = (const float*)d_in[16], *l1g2 = (const float*)d_in[17], *l1b2 = (const float*)d_in[18];
  const float* ptl1_w = (const float*)d_in[19], *ptl1_g = (const float*)d_in[20], *ptl1_b = (const float*)d_in[21];
  const float* ptl2_w = (const float*)d_in[22], *ptl2_g = (const float*)d_in[23], *ptl2_b = (const float*)d_in[24];
  // sa i: base 25 + 7i : w_qk, w_v, b_v, w_t, b_t, bn_g, bn_b
  const float* fuse_w = (const float*)d_in[53];
  const float* fuse_g = (const float*)d_in[54];
  const float* fuse_b = (const float*)d_in[55];

  // ---- workspace carve (assumes ws_size >= ~1.2 GB; trivial vs 432 GiB HBM) ----
  size_t off = 0;
  auto carve = [&](size_t bytes) -> void* {
    void* p = (char*)d_ws + off;
    off += (bytes + 255) & ~(size_t)255;
    return p;
  };
  float* big0 = (float*)carve((size_t)536870912);              // 128 Mi floats
  float* big1 = (float*)carve((size_t)536870912);
  float* h1   = (float*)carve((size_t)Bn * 64 * Np * 4);
  float* h2   = (float*)carve((size_t)Bn * 64 * Np * 4);
  float* f0   = (float*)carve((size_t)Bn * 128 * S0 * 4);
  float* nx0  = (float*)carve((size_t)Bn * 3 * S0 * 4);
  int*   fi0  = (int*)carve((size_t)Bn * S0 * 4);
  int*   fi1  = (int*)carve((size_t)Bn * S1 * 4);
  int*   kn0  = (int*)carve((size_t)Bn * S0 * 32 * 4);
  int*   kn1  = (int*)carve((size_t)Bn * S1 * 32 * 4);
  float* cat  = (float*)carve((size_t)Bn * 1280 * 256 * 4);
  float* sums = (float*)carve((size_t)2 * 1024 * 4);
  float* meanB = (float*)carve((size_t)1024 * 4);
  float* rstdB = (float*)carve((size_t)1024 * 4);
  // Small SA-stage buffers overlaid onto big0 (free after the local stages).
  const long M2 = 256L * 256;
  float* hA    = big0;
  float* hB    = hA + (long)Bn * M2;
  float* yqk   = hB + (long)Bn * M2;
  float* Ebuf  = yqk + (long)Bn * 64 * 256;
  float* vbuf  = Ebuf + (long)Bn * M2;
  float* xrbuf = vbuf + (long)Bn * M2;
  float* zbuf  = xrbuf + (long)Bn * M2;
  float* tbuf  = zbuf + (long)Bn * M2;
  float* fused = tbuf + (long)Bn * M2;     // [B,1024,256]

  auto GEMM = [&](const float* A, long aBS, int lda, int aT,
                  const float* Bp, long bBS, int ldb,
                  float* Cp, long cBS, int ldc, int M, int Nc, int Kd,
                  const float* bias) {
    dim3 g((Nc + 63) / 64, (M + 63) / 64, Bn);
    gemm_wmma<<<g, 128, 0, stream>>>(A, aBS, lda, aT, Bp, bBS, ldb, Cp, cBS, ldc, M, Nc, Kd, bias);
  };
  auto BN = [&](const float* Y, int Cc, long Nn, const float* g, const float* bt,
                const float* res, long resBS, float* Out, long outBS, int act) {
    long perCh = (long)Bn * Nn;
    zero_f32<<<(2 * Cc + 255) / 256, 256, 0, stream>>>(sums, 2L * Cc);
    int slabs = 32;
    dim3 gs(Cc, slabs);
    bn_partial<<<gs, 256, 0, stream>>>(Y, Cc, Nn, perCh, slabs, sums);
    bn_finalize<<<(Cc + 255) / 256, 256, 0, stream>>>(sums, Cc, (float)perCh, 1e-5f, meanB, rstdB);
    long tot = (long)Bn * Cc * Nn;
    bn_apply<<<(tot + 255) / 256, 256, 0, stream>>>(Y, meanB, rstdB, g, bt, res, resBS, Out, outBS, Cc, Nn, tot, act);
  };

  // ---- stem convs: h2 = relu(bn(W2 * relu(bn(W1 * x)))) ----
  GEMM(conv1_w, 0, 3, 0, x, 3L * Np, Np, h1, 64L * Np, Np, 64, Np, 3, nullptr);
  BN(h1, 64, Np, bn1_g, bn1_b, nullptr, 0, h1, 64L * Np, 1);
  GEMM(conv2_w, 0, 64, 0, h1, 64L * Np, Np, h2, 64L * Np, Np, 64, Np, 64, nullptr);
  BN(h2, 64, Np, bn2_g, bn2_b, nullptr, 0, h2, 64L * Np, 1);

  // ---- FPS / KNN / group (stage 0) ----
  fps_kernel<4><<<Bn, 256, 0, stream>>>(x, Np, S0, fi0);
  gather_xyz<<<((long)Bn * 3 * S0 + 255) / 256, 256, 0, stream>>>(x, fi0, Np, S0, (long)Bn * 3 * S0, nx0);
  knn_kernel<1024><<<dim3(S0, Bn), 256, 0, stream>>>(x, fi0, Np, S0, kn0);
  const long cols0 = (long)S0 * 32;
  gather_group<<<((long)Bn * 64 * cols0 + 255) / 256, 256, 0, stream>>>(
      h2, fi0, kn0, Np, S0, 64, (long)Bn * 64 * cols0, big0);

  // ---- local op 0: [B,128,16384] ----
  GEMM(l0w1, 0, 128, 0, big0, 128L * cols0, (int)cols0, big1, 128L * cols0, (int)cols0, 128, (int)cols0, 128, nullptr);
  BN(big1, 128, cols0, l0g1, l0b1, nullptr, 0, big1, 128L * cols0, 1);
  GEMM(l0w2, 0, 128, 0, big1, 128L * cols0, (int)cols0, big0, 128L * cols0, (int)cols0, 128, (int)cols0, 128, nullptr);
  BN(big0, 128, cols0, l0g2, l0b2, nullptr, 0, big0, 128L * cols0, 1);
  max_over_k<<<((long)Bn * 128 * S0 + 255) / 256, 256, 0, stream>>>(
      big0, S0, 128, 128L * S0, (long)Bn * 128 * S0, f0);

  // ---- FPS / KNN / group (stage 1) ----
  fps_kernel<2><<<Bn, 256, 0, stream>>>(nx0, S0, S1, fi1);
  knn_kernel<512><<<dim3(S1, Bn), 256, 0, stream>>>(nx0, fi1, S0, S1, kn1);
  const long cols1 = (long)S1 * 32;
  gather_group<<<((long)Bn * 128 * cols1 + 255) / 256, 256, 0, stream>>>(
      f0, fi1, kn1, S0, S1, 128, (long)Bn * 128 * cols1, big1);

  // ---- local op 1: [B,256,8192]; f1 written into cat rows 1024..1279 ----
  GEMM(l1w1, 0, 256, 0, big1, 256L * cols1, (int)cols1, big0, 256L * cols1, (int)cols1, 256, (int)cols1, 256, nullptr);
  BN(big0, 256, cols1, l1g1, l1b1, nullptr, 0, big0, 256L * cols1, 1);
  GEMM(l1w2, 0, 256, 0, big0, 256L * cols1, (int)cols1, big1, 256L * cols1, (int)cols1, 256, (int)cols1, 256, nullptr);
  BN(big1, 256, cols1, l1g2, l1b2, nullptr, 0, big1, 256L * cols1, 1);
  float* f1 = cat + 1024L * 256;   // per-batch stride 1280*256
  max_over_k<<<((long)Bn * 256 * S1 + 255) / 256, 256, 0, stream>>>(
      big1, S1, 256, 1280L * 256, (long)Bn * 256 * S1, f1);

  // ---- point transformer-last convs ----
  GEMM(ptl1_w, 0, 256, 0, f1, 1280L * 256, 256, hA, M2, 256, 256, 256, 256, nullptr);
  BN(hA, 256, 256, ptl1_g, ptl1_b, nullptr, 0, hA, M2, 1);
  GEMM(ptl2_w, 0, 256, 0, hA, M2, 256, hB, M2, 256, 256, 256, 256, nullptr);
  BN(hB, 256, 256, ptl2_g, ptl2_b, nullptr, 0, hB, M2, 1);

  // ---- 4x offset-attention SA layers; outputs placed into cat slices ----
  const float* xin = hB; long xinBS = M2;
  for (int i = 0; i < 4; ++i) {
    const float* wqk = (const float*)d_in[25 + 7 * i + 0];
    const float* wv  = (const float*)d_in[25 + 7 * i + 1];
    const float* bv  = (const float*)d_in[25 + 7 * i + 2];
    const float* wt  = (const float*)d_in[25 + 7 * i + 3];
    const float* bt  = (const float*)d_in[25 + 7 * i + 4];
    const float* bg  = (const float*)d_in[25 + 7 * i + 5];
    const float* bb  = (const float*)d_in[25 + 7 * i + 6];
    float* xout = cat + (long)i * 256 * 256;

    // y = W_qk * x  -> energy = y^T y (Gram matrix; q and k share weights)
    GEMM(wqk, 0, 256, 0, xin, xinBS, 256, yqk, 64L * 256, 256, 64, 256, 256, nullptr);
    GEMM(yqk, 64L * 256, 256, 1, yqk, 64L * 256, 256, Ebuf, M2, 256, 256, 256, 64, nullptr);
    softmax_rows<<<Bn * 256, 256, 0, stream>>>(Ebuf, 256);
    att_colnorm<<<Bn, 256, 0, stream>>>(Ebuf, 256);
    // v = W_v x + b_v ; x_r = v * att
    GEMM(wv, 0, 256, 0, xin, xinBS, 256, vbuf, M2, 256, 256, 256, 256, bv);
    GEMM(vbuf, M2, 256, 0, Ebuf, M2, 256, xrbuf, M2, 256, 256, 256, 256, nullptr);
    // t = W_t (x - x_r) + b_t ; out = x + relu(bn(t))
    ew_sub<<<((long)Bn * M2 + 255) / 256, 256, 0, stream>>>(xin, xinBS, xrbuf, M2, Bn, zbuf);
    GEMM(wt, 0, 256, 0, zbuf, M2, 256, tbuf, M2, 256, 256, 256, 256, bt);
    BN(tbuf, 256, 256, bg, bb, xin, xinBS, xout, 1280L * 256, 1);

    xin = xout; xinBS = 1280L * 256;
  }

  // ---- fuse: [1024,1280] x cat[B,1280,256] -> bn -> leaky(0.2) -> max over N ----
  GEMM(fuse_w, 0, 1280, 0, cat, 1280L * 256, 256, fused, 1024L * 256, 256, 1024, 256, 1280, nullptr);
  BN(fused, 1024, 256, fuse_g, fuse_b, nullptr, 0, fused, 1024L * 256, 2);
  max_over_n<<<((long)Bn * 1024 + 255) / 256, 256, 0, stream>>>(
      fused, 256, (long)Bn * 1024, (float*)d_out);
}
